// Decoder_65051574665713
// MI455X (gfx1250) — matrix operand order repaired and checked by reading, not compiled, not run on hardware
//
#include <hip/hip_runtime.h>
#include <math.h>

// ---------------- problem constants ----------------
#define LNUM   12
#define HNUM   12
#define CDIM   768
#define DH     64
#define BDIM   4
#define TTOK   64
#define TIMG   256
#define TSEQ   320          // TIMG + TTOK
#define VDIM   1664
#define HDIM   896
#define VOCAB  50257
#define LORA_R 40
#define LORA_SC (1.0f/40.0f)

// ---------------- WMMA types (gfx1250, wave32) ----------------
typedef __attribute__((ext_vector_type(16))) __bf16 v16bf;
typedef __attribute__((ext_vector_type(8)))  __bf16 v8bf;
typedef __attribute__((ext_vector_type(8)))  float  v8f;

__device__ __forceinline__ v16bf make_frag(const __bf16* lo, const __bf16* hi) {
  v8bf a = *(const v8bf*)lo;                // ds_read_b128
  v8bf b = *(const v8bf*)hi;                // ds_read_b128
  v16bf r;
#pragma unroll
  for (int i = 0; i < 8; ++i) { r[i] = a[i]; r[i + 8] = b[i]; }
  return r;
}

// ---------------- generic bf16 WMMA GEMM ----------------
// out[m][n] = act(alpha*sum_k A[m][k]*W[n][k] + bias[n] + add0[m][n] + add1[m][n]) * colscale[n]
// All launches guarantee M % BM == 0 and K % 8 == 0; staged 8-elem chunks are all-or-nothing
// in bounds, so EDGE only needs chunk-level predicates (no per-element guards).
#define BM 128
#define BN 64
#define BK 32
#define ASTR 40   // LDS row stride in bf16 elems (80B = 20 words -> conflict-free, 16B aligned)

enum { ACT_NONE = 0, ACT_GELU_ERF = 1, ACT_GELU_TANH = 2 };

// noinline: transcendental bodies appear once in the binary; called once per block tile
__device__ __attribute__((noinline)) float gelu_erf_f(float v) {
  return 0.5f * v * (1.0f + erff(v * 0.70710678118f));
}
__device__ __attribute__((noinline)) float gelu_tanh_f(float v) {
  return 0.5f * v * (1.0f + tanhf(0.7978845608f * (v + 0.044715f * v * v * v)));
}

template <int ACT>
__device__ __forceinline__ float epilogue_val(float acc, float alpha, float bias_v,
                                              const float* add0, const float* add1,
                                              size_t o, float cs) {
  float v = alpha * acc + bias_v;
  if (add0) v += add0[o];
  if (add1) v += add1[o];
  if (ACT == ACT_GELU_ERF)  v = gelu_erf_f(v);
  if (ACT == ACT_GELU_TANH) v = gelu_tanh_f(v);
  return v * cs;
}

template <bool EDGE>
__device__ __forceinline__ void glload(const float* p, bool ok, float4* f) {
  if (!EDGE || ok) {
    f[0] = ((const float4*)p)[0];   // global_load_b128
    f[1] = ((const float4*)p)[1];   // global_load_b128
  } else {
    f[0] = make_float4(0.f, 0.f, 0.f, 0.f);
    f[1] = make_float4(0.f, 0.f, 0.f, 0.f);
  }
}

__device__ __forceinline__ void store8(__bf16* dst, const float4* f) {
  v8bf hv;                           // native cvt -> v_cvt_pk_bf16_f32
  hv[0] = (__bf16)f[0].x; hv[1] = (__bf16)f[0].y;
  hv[2] = (__bf16)f[0].z; hv[3] = (__bf16)f[0].w;
  hv[4] = (__bf16)f[1].x; hv[5] = (__bf16)f[1].y;
  hv[6] = (__bf16)f[1].z; hv[7] = (__bf16)f[1].w;
  *(v8bf*)dst = hv;                  // ds_store_b128
}

template <bool EDGE, int ACT>
__global__ __launch_bounds__(256)
void gemm_bf16_wmma(const float* __restrict__ A, const float* __restrict__ W,
                    const float* __restrict__ bias,
                    const float* add0, const float* add1,
                    const float* __restrict__ colscale,
                    float* out, int M, int N, int K, float alpha) {
  __shared__ __bf16 As[BM * ASTR];   // 128x32 bf16, stride 40
  __shared__ __bf16 Ws[BN * ASTR];   //  64x32 bf16, stride 40

  const int tid  = threadIdx.x;
  const int lane = tid & 31;
  const int wid  = tid >> 5;            // 8 waves
  const int wm   = wid >> 2;            // 0..1  (64-row strips)
  const int wn   = wid & 3;             // 0..3  (16-col strips)
  const int m0   = blockIdx.y * BM;
  const int n0   = blockIdx.x * BN;

  // staging coordinates: A = 512 chunks (2/thread), W = 256 chunks (1/thread)
  const int ra0 = tid >> 2;             // A rows 0..63
  const int ra1 = ra0 + 64;             // A rows 64..127
  const int kc  = (tid & 3) << 3;       // 0,8,16,24 within k-tile
  const float* pa0 = A + (size_t)(m0 + ra0) * K + kc;
  const float* pa1 = A + (size_t)(m0 + ra1) * K + kc;
  const float* pw  = W + (size_t)(n0 + ra0) * K + kc;
  const bool wrow_ok = !EDGE || (n0 + ra0) < N;

  v8f acc[4] = {};                      // four 16x16 tiles stacked in M per wave
  const int ktiles = (K + BK - 1) / BK;

  // software pipeline: preload k-tile 0 into registers
  float4 fa0[2], fa1[2], fw[2];
  {
    const bool kok = !EDGE || (kc + 8) <= K;
    glload<EDGE>(pa0, kok, fa0);
    glload<EDGE>(pa1, kok, fa1);
    glload<EDGE>(pw, wrow_ok && kok, fw);
  }

  for (int kt = 0; kt < ktiles; ++kt) {
    // convert held registers -> LDS
    store8(&As[ra0 * ASTR + kc], fa0);
    store8(&As[ra1 * ASTR + kc], fa1);
    store8(&Ws[ra0 * ASTR + kc], fw);
    __syncthreads();

    // issue next k-tile's global loads; latency overlaps the WMMA phase below
    if (kt + 1 < ktiles) {
      pa0 += BK; pa1 += BK; pw += BK;
      const bool kok = !EDGE || ((kt + 1) * BK + kc + 8) <= K;
      glload<EDGE>(pa0, kok, fa0);
      glload<EDGE>(pa1, kok, fa1);
      glload<EDGE>(pw, wrow_ok && kok, fw);
      if (!EDGE) __builtin_prefetch(pw + BK, 0, 1);   // global_prefetch_b8 for tile kt+2
    }

    // B fragment 32x16 (contiguous k, same order as A): N = lane%16, K = (lane<16 ? 0 : 16) + i
    const int hl = lane >> 4;
    const __bf16* br = &Ws[(wn * 16 + (lane & 15)) * ASTR + (hl ? 16 : 0)];
    const v16bf bf = make_frag(br, br + 8);

    // A fragments 16x32: lanes<16 -> K 0..15 ; lanes>=16 -> K 16..31
#pragma unroll
    for (int sub = 0; sub < 4; ++sub) {
      const __bf16* ar = &As[(wm * 64 + sub * 16 + (lane & 15)) * ASTR];
      const v16bf a = make_frag(ar + (hl ? 16 : 0), ar + (hl ? 24 : 8));
      acc[sub] = __builtin_amdgcn_wmma_f32_16x16x32_bf16(false, a, false, bf,
                                                         (short)0, acc[sub], false, false);
    }
    __syncthreads();
  }

  // C/D layout: VGPR i, lanes<16 -> M=i ; lanes>=16 -> M=8+i ; N = lane%16
  const int gn = n0 + wn * 16 + (lane & 15);
  if (EDGE && gn >= N) return;
  const float bias_v = bias ? bias[gn] : 0.0f;
  const float cs     = colscale ? colscale[gn] : 1.0f;
#pragma unroll
  for (int sub = 0; sub < 4; ++sub) {
    const int rb = m0 + wm * 64 + sub * 16 + ((lane >> 4) << 3);
#pragma unroll
    for (int i = 0; i < 8; ++i) {
      const size_t o = (size_t)(rb + i) * N + gn;     // rows always in bounds (M % 128 == 0)
      out[o] = epilogue_val<ACT>(acc[sub][i], alpha, bias_v, add0, add1, o, cs);
    }
  }
}

// ---------------- LayerNorm (one block per row) ----------------
__device__ __forceinline__ void ln_body(const float* src, const float* w,
                                        const float* bb, float* dst, int C) {
  __shared__ float red[256];
  const int tid = threadIdx.x;
  float s = 0.0f;
  for (int c = tid; c < C; c += 256) s += src[c];
  red[tid] = s; __syncthreads();
  for (int o = 128; o > 0; o >>= 1) { if (tid < o) red[tid] += red[tid + o]; __syncthreads(); }
  const float mean = red[0] / (float)C;
  __syncthreads();
  float v = 0.0f;
  for (int c = tid; c < C; c += 256) { float d = src[c] - mean; v += d * d; }
  red[tid] = v; __syncthreads();
  for (int o = 128; o > 0; o >>= 1) { if (tid < o) red[tid] += red[tid + o]; __syncthreads(); }
  const float rstd = rsqrtf(red[0] / (float)C + 1e-5f);
  for (int c = tid; c < C; c += 256)
    dst[c] = (src[c] - mean) * rstd * w[c] + bb[c];
}

__global__ __launch_bounds__(256)
void ln_rows(const float* __restrict__ x, const float* __restrict__ w,
             const float* __restrict__ b, float* __restrict__ y, int C) {
  const size_t r = blockIdx.x;
  ln_body(x + r * C, w, b, y + r * C, C);
}

// final LN over text rows only: row r -> x[b*TSEQ + TIMG + t]
__global__ __launch_bounds__(256)
void lnf_gather(const float* __restrict__ x, const float* __restrict__ w,
                const float* __restrict__ b, float* __restrict__ xf) {
  const int r = blockIdx.x;          // 0..255
  const int bb = r >> 6, t = r & 63;
  ln_body(x + ((size_t)(bb * TSEQ + TIMG + t)) * CDIM, w, b, xf + (size_t)r * CDIM, CDIM);
}

// ---------------- embedding / vision plumbing ----------------
__global__ __launch_bounds__(256)
void embed_kernel(const int* __restrict__ tok, const float* __restrict__ wte,
                  const float* __restrict__ wpe, const float* __restrict__ gate,
                  float* __restrict__ x) {
  const int blk = blockIdx.x;        // B*TTOK
  const int b = blk >> 6, t = blk & 63;
  const int tk = tok[b * TTOK + t];
  float* dst = x + ((size_t)(b * TSEQ + TIMG + t)) * CDIM;
  const float* we = wte + (size_t)tk * CDIM;
  const float* pe = wpe + (size_t)t * CDIM;
  for (int c = threadIdx.x; c < CDIM; c += 256) dst[c] = (we[c] + pe[c]) * gate[c];
}

__global__ __launch_bounds__(256)
void extract_vis(const float* __restrict__ imf, float* __restrict__ vis) {
  const int r = blockIdx.x;          // B*TIMG
  const int b = r >> 8, i = r & 255;
  const float* src = imf + ((size_t)(b * (TIMG + 1) + i + 1)) * VDIM;  // skip CLS
  float* dst = vis + (size_t)r * VDIM;
  for (int c = threadIdx.x; c < VDIM; c += 256) dst[c] = src[c];
}

__global__ __launch_bounds__(256)
void scatter_proj(const float* __restrict__ proj, float* __restrict__ x) {
  const int r = blockIdx.x;          // B*TIMG
  const int b = r >> 8, i = r & 255;
  const float* src = proj + (size_t)r * CDIM;
  float* dst = x + ((size_t)(b * TSEQ + i)) * CDIM;
  for (int c = threadIdx.x; c < CDIM; c += 256) dst[c] = src[c];
}

// ---------------- causal softmax attention (one block per b,h,t) ----------------
__global__ __launch_bounds__(64)
void attn_softmax(const float* __restrict__ qkv, float* __restrict__ o) {
  const int idx = blockIdx.x;
  const int t = idx % TSEQ;
  const int h = (idx / TSEQ) % HNUM;
  const int b = idx / (TSEQ * HNUM);
  const int tid = threadIdx.x;       // 64 threads
  const int C3 = 3 * CDIM;

  __shared__ float qs[DH];
  __shared__ float sc[TSEQ];
  __shared__ float red[64];

  const float* qrow = qkv + ((size_t)(b * TSEQ + t)) * C3 + h * DH;
  if (tid < DH) qs[tid] = qrow[tid];
  __syncthreads();

  float lmax = -3.0e38f;
  for (int s = tid; s <= t; s += 64) {
    const float* krow = qkv + ((size_t)(b * TSEQ + s)) * C3 + CDIM + h * DH;
    float d = 0.0f;
#pragma unroll 16
    for (int j = 0; j < DH; ++j) d += qs[j] * krow[j];
    d *= 0.125f;                      // 1/sqrt(64)
    sc[s] = d;
    lmax = fmaxf(lmax, d);
  }
  red[tid] = lmax; __syncthreads();
  for (int off = 32; off > 0; off >>= 1) { if (tid < off) red[tid] = fmaxf(red[tid], red[tid + off]); __syncthreads(); }
  const float mx = red[0];
  __syncthreads();

  float lsum = 0.0f;
  for (int s = tid; s <= t; s += 64) { float e = __expf(sc[s] - mx); sc[s] = e; lsum += e; }
  red[tid] = lsum; __syncthreads();
  for (int off = 32; off > 0; off >>= 1) { if (tid < off) red[tid] += red[tid + off]; __syncthreads(); }
  const float inv = 1.0f / red[0];

  if (tid < DH) {
    float acc = 0.0f;
    const float* vbase = qkv + ((size_t)(b * TSEQ)) * C3 + 2 * CDIM + h * DH + tid;
    for (int s = 0; s <= t; ++s) acc += sc[s] * vbase[(size_t)s * C3];
    o[((size_t)(b * TSEQ + t)) * CDIM + h * DH + tid] = acc * inv;
  }
}

// ---------------- host orchestration ----------------
extern "C" void kernel_launch(void* const* d_in, const int* in_sizes, int n_in,
                              void* d_out, int out_size, void* d_ws, size_t ws_size,
                              hipStream_t stream) {
  (void)in_sizes; (void)n_in; (void)out_size; (void)ws_size;

  const float* image_features = (const float*)d_in[0];
  const int*   text_tokens    = (const int*)  d_in[1];
  const float* wte    = (const float*)d_in[2];
  const float* wpe    = (const float*)d_in[3];
  const float* vp_w1  = (const float*)d_in[4];
  const float* vp_b1  = (const float*)d_in[5];
  const float* vp_w2  = (const float*)d_in[6];
  const float* vp_b2  = (const float*)d_in[7];
  const float* gate_image = (const float*)d_in[8];
  const float* gate_text  = (const float*)d_in[9];
  const float* ln1_w  = (const float*)d_in[10];
  const float* ln1_b  = (const float*)d_in[11];
  const float* attn_w = (const float*)d_in[12];
  const float* attn_b = (const float*)d_in[13];
  const float* attn_A = (const float*)d_in[14];
  const float* attn_B = (const float*)d_in[15];
  const float* proj_w = (const float*)d_in[16];
  const float* proj_b = (const float*)d_in[17];
  const float* proj_A = (const float*)d_in[18];
  const float* proj_B = (const float*)d_in[19];
  const float* ln2_w  = (const float*)d_in[20];
  const float* ln2_b  = (const float*)d_in[21];
  const float* fc_w   = (const float*)d_in[22];
  const float* fc_b   = (const float*)d_in[23];
  const float* mproj_w = (const float*)d_in[24];
  const float* mproj_b = (const float*)d_in[25];
  const float* lnf_w  = (const float*)d_in[26];
  const float* lnf_b  = (const float*)d_in[27];

  const int Mrows = BDIM * TSEQ;     // 1280

  char* ws = (char*)d_ws;
  size_t off = 0;
  auto carve = [&](size_t nflt) {
    float* p = (float*)(ws + off);
    off += ((nflt * sizeof(float)) + 255) & ~(size_t)255;
    return p;
  };
  float* x    = carve((size_t)Mrows * CDIM);
  float* y    = carve((size_t)Mrows * CDIM);
  float* qkv  = carve((size_t)Mrows * 3 * CDIM);
  float* lmid = carve((size_t)Mrows * 64);
  float* obuf = carve((size_t)Mrows * CDIM);
  float* o2   = carve((size_t)Mrows * CDIM);
  float* xf   = carve((size_t)BDIM * TTOK * CDIM);
  float* m1   = carve((size_t)Mrows * 4 * CDIM);        // 15.7 MB
  // vision temps live inside the (not-yet-used) m1 region
  float* vis  = m1;                                     // 1024*1664
  float* hbuf = m1 + (size_t)BDIM * TIMG * VDIM;        // 1024*896
  float* pbuf = o2;                                     // 1024*768 temp

  dim3 blk(256);
  auto gemm = [&](const float* A, const float* W, const float* bias,
                  const float* a0, const float* a1, const float* cs,
                  float* outp, int M, int N, int K, float alpha, int act, bool edge) {
    dim3 g((unsigned)((N + BN - 1) / BN), (unsigned)((M + BM - 1) / BM));
    if (edge)         // edge calls always ACT_NONE in this model
      gemm_bf16_wmma<true,  ACT_NONE><<<g, blk, 0, stream>>>(A, W, bias, a0, a1, cs, outp, M, N, K, alpha);
    else if (act == ACT_GELU_ERF)
      gemm_bf16_wmma<false, ACT_GELU_ERF><<<g, blk, 0, stream>>>(A, W, bias, a0, a1, cs, outp, M, N, K, alpha);
    else if (act == ACT_GELU_TANH)
      gemm_bf16_wmma<false, ACT_GELU_TANH><<<g, blk, 0, stream>>>(A, W, bias, a0, a1, cs, outp, M, N, K, alpha);
    else
      gemm_bf16_wmma<false, ACT_NONE><<<g, blk, 0, stream>>>(A, W, bias, a0, a1, cs, outp, M, N, K, alpha);
  };

  // --- embedding + vision projector ---
  embed_kernel<<<BDIM * TTOK, 256, 0, stream>>>(text_tokens, wte, wpe, gate_text, x);
  extract_vis<<<BDIM * TIMG, 256, 0, stream>>>(image_features, vis);
  gemm(vis,  vp_w1, vp_b1, nullptr, nullptr, nullptr,    hbuf, BDIM * TIMG, HDIM, VDIM, 1.0f, ACT_GELU_ERF, false);
  gemm(hbuf, vp_w2, vp_b2, nullptr, nullptr, gate_image, pbuf, BDIM * TIMG, CDIM, HDIM, 1.0f, ACT_NONE, false);
  scatter_proj<<<BDIM * TIMG, 256, 0, stream>>>(pbuf, x);

  // --- transformer layers ---
  for (int l = 0; l < LNUM; ++l) {
    ln_rows<<<Mrows, 256, 0, stream>>>(x, ln1_w + (size_t)l * CDIM, ln1_b + (size_t)l * CDIM, y, CDIM);
    gemm(y, attn_w + (size_t)l * 3 * CDIM * CDIM, attn_b + (size_t)l * 3 * CDIM,
         nullptr, nullptr, nullptr, qkv, Mrows, 3 * CDIM, CDIM, 1.0f, ACT_NONE, false);
    gemm(y, attn_A + (size_t)l * LORA_R * CDIM, nullptr,
         nullptr, nullptr, nullptr, lmid, Mrows, LORA_R, CDIM, 1.0f, ACT_NONE, true);
    gemm(lmid, attn_B + (size_t)l * 3 * CDIM * LORA_R, nullptr,
         qkv, nullptr, nullptr, qkv, Mrows, 3 * CDIM, LORA_R, LORA_SC, ACT_NONE, true);

    attn_softmax<<<BDIM * HNUM * TSEQ, 64, 0, stream>>>(qkv, obuf);

    gemm(obuf, proj_w + (size_t)l * CDIM * CDIM, proj_b + (size_t)l * CDIM,
         nullptr, nullptr, nullptr, o2, Mrows, CDIM, CDIM, 1.0f, ACT_NONE, false);
    gemm(obuf, proj_A + (size_t)l * LORA_R * CDIM, nullptr,
         nullptr, nullptr, nullptr, lmid, Mrows, LORA_R, CDIM, 1.0f, ACT_NONE, true);
    // x = x + o2 + lora*scale   (fused residual)
    gemm(lmid, proj_B + (size_t)l * CDIM * LORA_R, nullptr,
         o2, x, nullptr, x, Mrows, CDIM, LORA_R, LORA_SC, ACT_NONE, true);

    ln_rows<<<Mrows, 256, 0, stream>>>(x, ln2_w + (size_t)l * CDIM, ln2_b + (size_t)l * CDIM, y, CDIM);
    gemm(y, fc_w + (size_t)l * 4 * CDIM * CDIM, fc_b + (size_t)l * 4 * CDIM,
         nullptr, nullptr, nullptr, m1, Mrows, 4 * CDIM, CDIM, 1.0f, ACT_GELU_TANH, false);
    gemm(m1, mproj_w + (size_t)l * CDIM * 4 * CDIM, mproj_b + (size_t)l * CDIM,
         x, nullptr, nullptr, x, Mrows, CDIM, 4 * CDIM, 1.0f, ACT_NONE, false);
  }

  // --- final LN on text rows + logits ---
  lnf_gather<<<BDIM * TTOK, 256, 0, stream>>>(x, lnf_w, lnf_b, xf);
  gemm(xf, wte, nullptr, nullptr, nullptr, nullptr,
       (float*)d_out, BDIM * TTOK, VOCAB, CDIM, 1.0f, ACT_NONE, true);
}
